// BackboneCNN_16836271801156
// MI455X (gfx1250) — compile-verified, matrix operand order checked by reading, not run on hardware
//
#include <hip/hip_runtime.h>
#include <hip/hip_bf16.h>

typedef __attribute__((ext_vector_type(16))) __bf16 v16bf;
typedef __attribute__((ext_vector_type(8)))  float  v8f;

__device__ __forceinline__ v8f wmma_bf16(v16bf a, v16bf b, v8f c) {
  // D = A(16x32 bf16) * B(32x16 bf16) + C(16x16 f32)
  return __builtin_amdgcn_wmma_f32_16x16x32_bf16(false, a, false, b, (short)0, c,
                                                 false, false);
}

__device__ __forceinline__ int reflect_idx(int i, int n) {
  if (i < 0) i = -i;
  if (i >= n) i = 2 * n - 2 - i;
  return i;
}

// ---------------------------------------------------------------------------
// Tensor Data Mover: 2D tile (f32) global -> LDS. D# built per CDNA5 ISA
// CDNA5 ISA (group0: count/lds_addr/global_addr/type; group1:
// data_size, tensor dims, tile dims, dim0 stride). Guarded by __has_builtin
// with 5-arg (ROCm7.2/clang-22) vs 6-arg (clang-23) arity switch.
// ---------------------------------------------------------------------------
#if __has_builtin(__builtin_amdgcn_tensor_load_to_lds)
#define HAS_TDM 1
#else
#define HAS_TDM 0
#endif

#if HAS_TDM
typedef unsigned int u32x4 __attribute__((ext_vector_type(4)));
typedef int i32x4 __attribute__((ext_vector_type(4)));
typedef int i32x8 __attribute__((ext_vector_type(8)));

__device__ __forceinline__ void tdm_load_2d(unsigned lds_off, const void* gptr,
                                            int tile_w, int tile_h,
                                            int tensor_w, int tensor_h,
                                            int row_stride_elts) {
  unsigned long long ga = (unsigned long long)(size_t)gptr;
  u32x4 g0 = {0u, 0u, 0u, 0u};
  g0[0] = 1u;                                            // count=1, user D#
  g0[1] = lds_off;                                       // lds_addr[31:0]
  g0[2] = (unsigned)ga;                                  // global_addr[31:0]
  g0[3] = (unsigned)((ga >> 32) & 0x01ffffffull) | (2u << 30); // addr[56:32]|type=2
  i32x8 g1 = {0, 0, 0, 0, 0, 0, 0, 0};
  g1[0] = 2 << 16;                                       // data_size = 4 bytes
  g1[1] = (tensor_w & 0xffff) << 16;                     // tensor_dim0 lo
  g1[2] = ((unsigned)tensor_w >> 16) | ((tensor_h & 0xffff) << 16);
  g1[3] = ((unsigned)tensor_h >> 16) | ((tile_w & 0xffff) << 16); // tile_dim0
  g1[4] = tile_h & 0xffff;                               // tile_dim1 (dim2=0)
  g1[5] = row_stride_elts;                               // tensor_dim0_stride lo
  i32x4 z4 = {0, 0, 0, 0};
#if defined(__clang_major__) && (__clang_major__ >= 23)
  i32x8 z8 = {0, 0, 0, 0, 0, 0, 0, 0};
  __builtin_amdgcn_tensor_load_to_lds(g0, g1, z4, z4, z8, 0);
#else
  __builtin_amdgcn_tensor_load_to_lds(g0, g1, z4, z4, 0);
#endif
}
#endif

// ---------------------------------------------------------------------------
// Mask-sum conv: msum over k*k window -> updated mask (d_out) + ratio (ws).
// ---------------------------------------------------------------------------
__global__ void mask_stage_kernel(const float* __restrict__ mprev,
                                  float* __restrict__ mout,
                                  float* __restrict__ ratio,
                                  int B, int Hi, int Wi, int Ho, int Wo,
                                  int k, int stride, int pad, int reflect) {
  int idx = blockIdx.x * blockDim.x + threadIdx.x;
  int P = Ho * Wo;
  if (idx >= B * P) return;
  int b = idx / P, p = idx - b * P;
  int oy = p / Wo, ox = p - oy * Wo;
  float s = 0.f;
  for (int r = 0; r < k; ++r) {
    for (int c = 0; c < k; ++c) {
      int iy = oy * stride - pad + r;
      int ix = ox * stride - pad + c;
      if (reflect) { iy = reflect_idx(iy, Hi); ix = reflect_idx(ix, Wi); }
      if (iy >= 0 && iy < Hi && ix >= 0 && ix < Wi)
        s += mprev[((size_t)b * Hi + iy) * Wi + ix];
    }
  }
  float win = (float)(k * k);
  float upd = (s / win > 0.5f) ? 1.f : 0.f;
  mout[idx] = upd;
  ratio[idx] = win / fmaxf(s, 1e-8f);
}

// ---------------------------------------------------------------------------
// Stage-1 partial conv (Ci=3, k=7, s=1, reflect pad 3) — all constants so the
// per-element K decode folds to magic multiplies.
// ---------------------------------------------------------------------------
__global__ __launch_bounds__(128) void pconv_s1_kernel(
    const float* __restrict__ x, const float* __restrict__ m,
    const float* __restrict__ w, const float* __restrict__ bias,
    const float* __restrict__ ratio, const float* __restrict__ upd,
    float* __restrict__ y) {
  constexpr int KSZ = 7, PAD = 3, CI = 3, H = 256, W = 256, CO = 64;
  constexpr int KK2 = KSZ * KSZ, K = CI * KK2;
  const int P = H * W;
  const int lane = threadIdx.x & 31;
  const int wave = threadIdx.x >> 5;
  const int p0 = (blockIdx.x * 4 + wave) * 16;
  const int co0 = blockIdx.y * 16;
  const int b = blockIdx.z;
  const int koff = (lane < 16) ? 0 : 8;
  const int mr = lane & 15;
  const int co = co0 + mr;
  const int p = p0 + mr;
  const int oy = p >> 8, ox = p & 255;
  v8f acc = {};
  for (int kb = 0; kb < K; kb += 32) {
    v16bf av, bv;
#pragma unroll
    for (int i = 0; i < 16; ++i) {
      int kkid = kb + koff + (i < 8 ? i : i + 8);
      float a = 0.f, vv = 0.f;
      if (kkid < K) {
        int ci = kkid / KK2;
        int rem = kkid - ci * KK2;
        int r = rem / KSZ, s = rem - r * KSZ;
        a = w[(co * CI + ci) * KK2 + rem];
        int iy = reflect_idx(oy - PAD + r, H);
        int ix = reflect_idx(ox - PAD + s, W);
        vv = x[(((size_t)b * CI + ci) * H + iy) * W + ix] *
             m[((size_t)b * H + iy) * W + ix];
      }
      av[i] = (__bf16)a;
      bv[i] = (__bf16)vv;
    }
    acc = wmma_bf16(av, bv, acc);
  }
  const float rt = ratio[(size_t)b * P + p];
  const float up = upd[(size_t)b * P + p];
#pragma unroll
  for (int v = 0; v < 8; ++v) {
    int oc = co0 + v + ((lane < 16) ? 0 : 8);
    y[((size_t)b * CO + oc) * P + p] = (acc[v] * rt + bias[oc]) * up;
  }
}

// ---------------------------------------------------------------------------
// Stages 2-4 partial conv: K permuted to (tap q, ci) so each 32-wide K chunk
// lives inside one window tap -> one mask load + constant-stride gathers.
// ---------------------------------------------------------------------------
template <int KSZ, int STR, int PAD, int CI>
__global__ __launch_bounds__(128) void pconv_qmajor_kernel(
    const float* __restrict__ x, const float* __restrict__ m,
    const float* __restrict__ w, const float* __restrict__ bias,
    const float* __restrict__ ratio, const float* __restrict__ upd,
    float* __restrict__ y, int Hi, int Wi, int Ho, int Wo, int Co) {
  constexpr int KK2 = KSZ * KSZ;
  const int P = Ho * Wo;
  const int lane = threadIdx.x & 31;
  const int wave = threadIdx.x >> 5;
  const int p0 = (blockIdx.x * 4 + wave) * 16;
  const int co0 = blockIdx.y * 16;
  const int b = blockIdx.z;
  const int koff = (lane < 16) ? 0 : 8;
  const int mr = lane & 15;
  const int co = co0 + mr;
  const int p = p0 + mr;
  const int oy = p / Wo, ox = p - (p / Wo) * Wo;
  const size_t chs = (size_t)Hi * Wi;          // channel stride in x
  v8f acc = {};
  for (int q = 0; q < KK2; ++q) {
    const int r = q / KSZ, s = q - (q / KSZ) * KSZ;
    const int iy = oy * STR - PAD + r;
    const int ix = ox * STR - PAD + s;
    const bool valid = (iy >= 0) & (iy < Hi) & (ix >= 0) & (ix < Wi);
    const float mval = valid ? m[((size_t)b * Hi + iy) * Wi + ix] : 0.f;
    const float* xb = x + (size_t)b * CI * chs + (size_t)iy * Wi + ix;
    const float* wb = w + (size_t)co * CI * KK2 + q;
#pragma unroll
    for (int cb = 0; cb < CI / 32; ++cb) {
      v16bf av, bv;
#pragma unroll
      for (int i = 0; i < 16; ++i) {
        const int ci = cb * 32 + koff + (i < 8 ? i : i + 8);
        av[i] = (__bf16)wb[(size_t)ci * KK2];
        bv[i] = (__bf16)(valid ? xb[(size_t)ci * chs] * mval : 0.f);
      }
      acc = wmma_bf16(av, bv, acc);
    }
  }
  const float rt = ratio[(size_t)b * P + p];
  const float up = upd[(size_t)b * P + p];
#pragma unroll
  for (int v = 0; v < 8; ++v) {
    int oc = co0 + v + ((lane < 16) ? 0 : 8);
    y[((size_t)b * Co + oc) * P + p] = (acc[v] * rt + bias[oc]) * up;
  }
}

// ---------------------------------------------------------------------------
// InstanceNorm(affine=False) + ReLU. One block per (b, c).
// ---------------------------------------------------------------------------
__global__ __launch_bounds__(256) void inorm_relu_kernel(
    const float* __restrict__ y, float* __restrict__ out, int P) {
  const size_t base = (size_t)blockIdx.x * P;
  __shared__ float s1[256], s2[256];
  float sum = 0.f, sq = 0.f;
  for (int i = threadIdx.x; i < P; i += 256) {
    float v = y[base + i];
    sum += v; sq += v * v;
  }
  s1[threadIdx.x] = sum; s2[threadIdx.x] = sq;
  __syncthreads();
  for (int off = 128; off > 0; off >>= 1) {
    if ((int)threadIdx.x < off) {
      s1[threadIdx.x] += s1[threadIdx.x + off];
      s2[threadIdx.x] += s2[threadIdx.x + off];
    }
    __syncthreads();
  }
  float mu = s1[0] / (float)P;
  float var = s2[0] / (float)P - mu * mu;
  float inv = rsqrtf(var + 1e-5f);
  for (int i = threadIdx.x; i < P; i += 256) {
    float v = (y[base + i] - mu) * inv;
    out[base + i] = v > 0.f ? v : 0.f;
  }
}

// ---------------------------------------------------------------------------
// Token projection memtok[b,n,:] = memory[b,n,:].lw^T + lb.
// lw row-block (16 x 512 f32, 32KB) staged to LDS by TDM, shared by 4 waves.
// ---------------------------------------------------------------------------
__global__ __launch_bounds__(128) void mem_proj_kernel(
    const float* __restrict__ memory, const float* __restrict__ lw,
    const float* __restrict__ lb, float* __restrict__ memtok,
    int NT, int D, int Co) {
  __shared__ float lw_lds[16 * 512];
  const int lane = threadIdx.x & 31;
  const int wave = threadIdx.x >> 5;
  const int n0 = (blockIdx.x * 4 + wave) * 16;
  const int co0 = blockIdx.y * 16;
  const int b = blockIdx.z;
#if HAS_TDM
  if (wave == 0) {
    tdm_load_2d((unsigned)(size_t)lw_lds, lw + (size_t)co0 * D,
                /*tile_w*/ D, /*tile_h*/ 16, /*tensor_w*/ D, /*tensor_h*/ Co,
                /*row_stride*/ D);
    __builtin_amdgcn_s_wait_tensorcnt(0);
  }
  __syncthreads();
#else
  for (int idx = threadIdx.x; idx < 16 * D; idx += 128)
    lw_lds[idx] = lw[(size_t)(co0 + idx / D) * D + (idx - (idx / D) * D)];
  __syncthreads();
#endif
  const int koff = (lane < 16) ? 0 : 8;
  const int n = n0 + (lane & 15);
  const int cl = lane & 15;
  v8f acc = {};
  for (int kb = 0; kb < D; kb += 32) {
    v16bf av, bv;
#pragma unroll
    for (int i = 0; i < 16; ++i) {
      int kk = kb + koff + (i < 8 ? i : i + 8);
      av[i] = (__bf16)memory[((size_t)b * NT + n) * D + kk];
      bv[i] = (__bf16)lw_lds[cl * D + kk];
    }
    acc = wmma_bf16(av, bv, acc);
  }
#pragma unroll
  for (int v = 0; v < 8; ++v) {
    int tn = n0 + v + ((lane < 16) ? 0 : 8);
    memtok[((size_t)b * NT + tn) * Co + co0 + cl] = acc[v] + lb[co0 + cl];
  }
}

__global__ void key_norm_kernel(const float* __restrict__ memtok,
                                float* __restrict__ knorm, int BN, int C) {
  int i = blockIdx.x * blockDim.x + threadIdx.x;
  if (i >= BN) return;
  const float* r = memtok + (size_t)i * C;
  float s = 0.f;
  for (int c = 0; c < C; ++c) { float v = r[c]; s += v * v; }
  knorm[i] = rsqrtf(s);
}

// ---------------------------------------------------------------------------
// Fused attention + mask blend (flash-style). One wave per 16-pixel tile.
// Q tile (C x 16, strided columns) staged to LDS by TDM; scores (16x256) live
// in LDS only; softmax in LDS; O = A.V via WMMA; epilogue blends.
// ---------------------------------------------------------------------------
template <int C_>
__global__ __launch_bounds__(32) void attn_blend_kernel(
    const float* __restrict__ xn, const float* __restrict__ mup,
    const float* __restrict__ memtok, const float* __restrict__ knorm,
    float* __restrict__ xnext, int P) {
  constexpr int NT = 256;
  const int lane = threadIdx.x & 31;
  const int p0 = blockIdx.x * 16;
  const int b = blockIdx.z;
  const int koff = (lane < 16) ? 0 : 8;
  const int mr = lane & 15;

  __shared__ float S[16 * NT];   // 16 KB scores
  __shared__ float Q[C_ * 16];   // <=16 KB Q tile, layout [c][pixel]

#if HAS_TDM
  tdm_load_2d((unsigned)(size_t)Q, xn + (size_t)b * C_ * P + p0,
              /*tile_w*/ 16, /*tile_h*/ C_, /*tensor_w*/ P, /*tensor_h*/ C_,
              /*row_stride*/ P);
  __builtin_amdgcn_s_wait_tensorcnt(0);
#else
  for (int idx = lane; idx < C_ * 16; idx += 32)
    Q[idx] = xn[((size_t)b * C_ + (idx >> 4)) * P + p0 + (idx & 15)];
#endif

  // ---- Phase 1: S = Q.K^T (keys = L2-normalized tokens) ---------------
  for (int nb = 0; nb < NT / 16; ++nb) {
    const int tok = nb * 16 + mr;
    const float kn = knorm[(size_t)b * NT + tok];
    const float* vrow = memtok + ((size_t)b * NT + tok) * C_;
    v8f acc = {};
    for (int kb = 0; kb < C_; kb += 32) {
      v16bf av, bv;
#pragma unroll
      for (int i = 0; i < 16; ++i) {
        int cc = kb + koff + (i < 8 ? i : i + 8);
        av[i] = (__bf16)Q[cc * 16 + mr];
        bv[i] = (__bf16)(vrow[cc] * kn);
      }
      acc = wmma_bf16(av, bv, acc);
    }
#pragma unroll
    for (int v = 0; v < 8; ++v) {
      int row = v + ((lane < 16) ? 0 : 8);
      S[row * NT + nb * 16 + mr] = acc[v];
    }
  }

  // ---- Phase 2: softmax over tokens, per pixel row --------------------
  if (lane < 16) {
    float mx = -1e30f;
    for (int n = 0; n < NT; ++n) mx = fmaxf(mx, S[lane * NT + n]);
    float sum = 0.f;
    for (int n = 0; n < NT; ++n) {
      float e = __expf(S[lane * NT + n] - mx);
      S[lane * NT + n] = e;
      sum += e;
    }
    float inv = 1.f / sum;
    for (int n = 0; n < NT; ++n) S[lane * NT + n] *= inv;
  }

  // ---- Phase 3: O = A.V, blend xnext = xn*m + O*(1-m) ------------------
  for (int cb = 0; cb < C_ / 16; ++cb) {
    const int c = cb * 16 + mr;
    v8f acc = {};
    for (int nb = 0; nb < NT / 32; ++nb) {
      v16bf av, bv;
#pragma unroll
      for (int i = 0; i < 16; ++i) {
        int t = nb * 32 + koff + (i < 8 ? i : i + 8);
        av[i] = (__bf16)S[mr * NT + t];
        bv[i] = (__bf16)memtok[((size_t)b * NT + t) * C_ + c];
      }
      acc = wmma_bf16(av, bv, acc);
    }
#pragma unroll
    for (int v = 0; v < 8; ++v) {
      int prow = v + ((lane < 16) ? 0 : 8);
      int pix = p0 + prow;
      float mm = mup[(size_t)b * P + pix];
      float xv = Q[c * 16 + prow];
      xnext[((size_t)b * C_ + c) * P + pix] = xv * mm + acc[v] * (1.f - mm);
    }
  }
}

// ---------------------------------------------------------------------------
// Host-side orchestration (graph-capture safe: launches only).
// ---------------------------------------------------------------------------
extern "C" void kernel_launch(void* const* d_in, const int* in_sizes, int n_in,
                              void* d_out, int out_size, void* d_ws, size_t ws_size,
                              hipStream_t stream) {
  const float* image  = (const float*)d_in[0];
  const float* maskin = (const float*)d_in[1];
  const float* memory = (const float*)d_in[2];
  const float* w[4]  = {(const float*)d_in[3],  (const float*)d_in[7],
                        (const float*)d_in[11], (const float*)d_in[15]};
  const float* bi[4] = {(const float*)d_in[4],  (const float*)d_in[8],
                        (const float*)d_in[12], (const float*)d_in[16]};
  const float* lw[4] = {(const float*)d_in[5],  (const float*)d_in[9],
                        (const float*)d_in[13], (const float*)d_in[17]};
  const float* lb[4] = {(const float*)d_in[6],  (const float*)d_in[10],
                        (const float*)d_in[14], (const float*)d_in[18]};

  const int B = 4, NT = 256, D = 512;
  const int Ci[4] = {3, 64, 128, 256};
  const int Co[4] = {64, 128, 256, 512};
  const int KK[4] = {7, 5, 3, 3};
  const int ST[4] = {1, 2, 2, 2};
  const int PD[4] = {3, 2, 1, 1};
  const int RF[4] = {1, 0, 0, 0};
  const int Hi[4] = {256, 256, 128, 64};
  const int Ho[4] = {256, 128, 64, 32};

  // d_out layout: x1,m1,x2,m2,x3,m3,x4,m4 flat
  float* out = (float*)d_out;
  float* xo[4]; float* mo[4];
  {
    size_t off = 0;
    for (int i = 0; i < 4; ++i) {
      size_t P = (size_t)Ho[i] * Ho[i];
      xo[i] = out + off; off += (size_t)B * Co[i] * P;
      mo[i] = out + off; off += (size_t)B * P;
    }
  }

  // workspace layout
  float* ws = (float*)d_ws;
  float* xin    = ws;                   // blended stage input   (<= 16.78M)
  float* yraw   = xin + 16777216;       // pre-norm conv output  (<= 16.78M)
  float* ratio  = yraw + 16777216;      // mask ratio            (<= 262144)
  float* memtok = ratio + 262144;       // projected tokens      (<= 262144)
  float* knorm  = memtok + 262144;      // key inv-norms         (1024)

  for (int i = 0; i < 4; ++i) {
    const int P = Ho[i] * Ho[i];
    const float* xsrc = (i == 0) ? image  : xin;
    const float* msrc = (i == 0) ? maskin : mo[i - 1];

    // 1) mask conv -> updated mask (d_out) + ratio (ws)
    mask_stage_kernel<<<(B * P + 255) / 256, 256, 0, stream>>>(
        msrc, mo[i], ratio, B, Hi[i], Hi[i], Ho[i], Ho[i],
        KK[i], ST[i], PD[i], RF[i]);

    // 2) partial conv (implicit GEMM, bf16 WMMA, f32 accumulate)
    dim3 cgrid(P / 64, Co[i] / 16, B);
    if (i == 0) {
      pconv_s1_kernel<<<cgrid, 128, 0, stream>>>(
          xsrc, msrc, w[i], bi[i], ratio, mo[i], yraw);
    } else if (i == 1) {
      pconv_qmajor_kernel<5, 2, 2, 64><<<cgrid, 128, 0, stream>>>(
          xsrc, msrc, w[i], bi[i], ratio, mo[i], yraw,
          Hi[i], Hi[i], Ho[i], Ho[i], Co[i]);
    } else if (i == 2) {
      pconv_qmajor_kernel<3, 2, 1, 128><<<cgrid, 128, 0, stream>>>(
          xsrc, msrc, w[i], bi[i], ratio, mo[i], yraw,
          Hi[i], Hi[i], Ho[i], Ho[i], Co[i]);
    } else {
      pconv_qmajor_kernel<3, 2, 1, 256><<<cgrid, 128, 0, stream>>>(
          xsrc, msrc, w[i], bi[i], ratio, mo[i], yraw,
          Hi[i], Hi[i], Ho[i], Ho[i], Co[i]);
    }

    // 3) instance norm + relu -> d_out (also serves as attention Q)
    inorm_relu_kernel<<<B * Co[i], 256, 0, stream>>>(yraw, xo[i], P);

    if (i < 3) {
      // 4) token projection (TDM-staged weights) + key norms
      mem_proj_kernel<<<dim3(NT / 64, Co[i] / 16, B), 128, 0, stream>>>(
          memory, lw[i], lb[i], memtok, NT, D, Co[i]);
      key_norm_kernel<<<(B * NT + 255) / 256, 256, 0, stream>>>(
          memtok, knorm, B * NT, Co[i]);
      // 5) fused attention + blend -> next stage input
      dim3 agrid(P / 16, 1, B);
      if (i == 0)
        attn_blend_kernel<64><<<agrid, 32, 0, stream>>>(
            xo[i], mo[i], memtok, knorm, xin, P);
      else if (i == 1)
        attn_blend_kernel<128><<<agrid, 32, 0, stream>>>(
            xo[i], mo[i], memtok, knorm, xin, P);
      else
        attn_blend_kernel<256><<<agrid, 32, 0, stream>>>(
            xo[i], mo[i], memtok, knorm, xin, P);
    }
  }
}
